// GCN_16303695855744
// MI455X (gfx1250) — compile-verified, matrix operand order checked by reading, not run on hardware
//
#include <hip/hip_runtime.h>

// GCN forward for MI455X (gfx1250, wave32).
// Edge scatter is the roofline (L2-resident fp32 atomics); dense transforms
// use v_wmma_f32_16x16x32_bf16 with weights pre-packed into WMMA B-layout.

typedef __attribute__((ext_vector_type(16))) __bf16 v16bf;
typedef __attribute__((ext_vector_type(8)))  float  v8f;

#define F_IN 256
#define HID   16
#define C_OUT 40

__device__ __forceinline__ void fadd(float* p, float v) {
#if defined(__gfx1250__)
    unsafeAtomicAdd(p, v);   // global_atomic_add_f32 (L2 atomic)
#else
    atomicAdd(p, v);
#endif
}

// K index inside a 32-wide chunk for packed-bf16 WMMA operands.
// v = VGPR pair index 0..7, khalf = lane>>4. Matches ISA 16-bit A/B layout.
__device__ __forceinline__ int wmma_k(int v, int khalf) {
    return ((v >> 2) * 16) + ((v & 3) * 2) + khalf * 8;
}

// ---------------- norm pipeline ----------------

__global__ void k_init_deg(float* deg, int n) {
    int i = blockIdx.x * blockDim.x + threadIdx.x;
    if (i < n) deg[i] = 1.0f;                 // self-loop weight
}

__global__ void k_deg_accum(const long long* __restrict__ ei,
                            const float* __restrict__ w,
                            float* deg, long long E) {
    long long e = blockIdx.x * (long long)blockDim.x + threadIdx.x;
    if (e < E) fadd(&deg[(int)ei[E + e]], w[e]);   // segment_sum over dst
}

__global__ void k_dinv(float* deg, int n) {
    int i = blockIdx.x * blockDim.x + threadIdx.x;
    if (i < n) {
        float d = deg[i];
        deg[i] = (d > 0.0f) ? rsqrtf(d) : 0.0f;
    }
}

__global__ void k_norm(const long long* __restrict__ ei,
                       const float* __restrict__ w,
                       const float* __restrict__ dinv,
                       float* __restrict__ norm, long long E) {
    long long e = blockIdx.x * (long long)blockDim.x + threadIdx.x;
    if (e < E) norm[e] = dinv[(int)ei[e]] * w[e] * dinv[(int)ei[E + e]];
}

__global__ void k_zero(float* p, long long n) {
    long long i = blockIdx.x * (long long)blockDim.x + threadIdx.x;
    if (i < n) p[i] = 0.0f;
}

// ---------------- weight pre-pack into WMMA B-layout (bf16) ----------------
// pw1: 8 chunks x 32 lanes x 16 bf16 (= full 256x16 W1).
__global__ void k_pack_w1(const float* __restrict__ W1, __bf16* __restrict__ pw1) {
    int tid = blockIdx.x * blockDim.x + threadIdx.x;
    if (tid >= 8 * 32) return;
    int kc = tid >> 5, lane = tid & 31;
    int m = lane & 15, khalf = lane >> 4;
    v16bf b;
    #pragma unroll
    for (int v = 0; v < 8; ++v) {
        int k = kc * 32 + wmma_k(v, khalf);
        b[2 * v]     = (__bf16)W1[k * HID + m];
        b[2 * v + 1] = (__bf16)W1[(k + 1) * HID + m];
    }
    ((v16bf*)pw1)[tid] = b;
}

// pw2: 3 n-tiles x 32 lanes x 16 bf16 (K padded 16->32, N padded 40->48).
__global__ void k_pack_w2(const float* __restrict__ W2, __bf16* __restrict__ pw2) {
    int tid = blockIdx.x * blockDim.x + threadIdx.x;
    if (tid >= 3 * 32) return;
    int t = tid >> 5, lane = tid & 31;
    int m = lane & 15, khalf = lane >> 4;
    int ncol = t * 16 + m;
    v16bf b;
    #pragma unroll
    for (int v = 0; v < 8; ++v) {
        int k = wmma_k(v, khalf);
        bool okc = (ncol < C_OUT);
        b[2 * v]     = (okc && k     < HID) ? (__bf16)W2[k * C_OUT + ncol]       : (__bf16)0.f;
        b[2 * v + 1] = (okc && k + 1 < HID) ? (__bf16)W2[(k + 1) * C_OUT + ncol] : (__bf16)0.f;
    }
    ((v16bf*)pw2)[tid] = b;
}

// ---------------- GEMM1: h1[N x 16] = x[N x 256] @ W1[256 x 16] ----------------
__global__ void k_gemm1(const float* __restrict__ x,
                        const __bf16* __restrict__ pw1,
                        float* __restrict__ h1, int n) {
    int wave = (blockIdx.x * blockDim.x + threadIdx.x) >> 5;
    int lane = threadIdx.x & 31;
    int row0 = wave * 16;
    if (row0 >= n) return;                    // wave-uniform; EXEC stays all-1s
    int m     = lane & 15;
    int khalf = lane >> 4;

    v8f acc = {0.f, 0.f, 0.f, 0.f, 0.f, 0.f, 0.f, 0.f};
    const float* arow = x + (long long)(row0 + m) * F_IN;
    const v16bf* bp   = (const v16bf*)pw1;

    #pragma unroll
    for (int kc = 0; kc < F_IN / 32; ++kc) {
        v16bf a;
        #pragma unroll
        for (int v = 0; v < 8; ++v) {
            int k = kc * 32 + wmma_k(v, khalf);
            a[2 * v]     = (__bf16)arow[k];
            a[2 * v + 1] = (__bf16)arow[k + 1];
        }
        v16bf b = bp[kc * 32 + lane];         // 32B aligned fragment load
        acc = __builtin_amdgcn_wmma_f32_16x16x32_bf16(
            false, a, false, b, (short)0, acc, false, false);
    }
    #pragma unroll
    for (int v = 0; v < 8; ++v)
        h1[(long long)(row0 + v + 8 * khalf) * HID + m] = acc[v];
}

// ---------------- scatter layer 1: one thread per edge, float4 gather ----------------
__global__ void k_scatter1(const long long* __restrict__ ei,
                           const float* __restrict__ norm,
                           const float* __restrict__ h1,
                           float* __restrict__ agg1, long long E) {
    long long e = blockIdx.x * (long long)blockDim.x + threadIdx.x;
    if (e >= E) return;
    int src = (int)ei[e];
    int dst = (int)ei[E + e];
    float nw = norm[e];
    const float4* hs = (const float4*)(h1 + (long long)src * HID);
    float*        ad = agg1 + (long long)dst * HID;
    #pragma unroll
    for (int q = 0; q < HID / 4; ++q) {
        float4 hv = hs[q];
        fadd(ad + 4 * q + 0, hv.x * nw);
        fadd(ad + 4 * q + 1, hv.y * nw);
        fadd(ad + 4 * q + 2, hv.z * nw);
        fadd(ad + 4 * q + 3, hv.w * nw);
    }
}

// ReLU + self-loop + bias: agg1 = relu(agg1 + h1*dinv^2 + b1)
__global__ void k_relu_self(const float* __restrict__ h1,
                            const float* __restrict__ dinv,
                            const float* __restrict__ b1,
                            float* __restrict__ agg1, int n) {
    long long i = blockIdx.x * (long long)blockDim.x + threadIdx.x;
    if (i >= (long long)n * HID) return;
    int node = (int)(i >> 4);
    int f    = (int)(i & 15);
    float di = dinv[node];
    float v  = agg1[i] + h1[i] * di * di + b1[f];
    agg1[i]  = v > 0.0f ? v : 0.0f;
}

// ---------------- GEMM2: t2[N x 40] = agg1[N x 16] @ W2[16 x 40] ----------------
__global__ void k_gemm2(const float* __restrict__ h,
                        const __bf16* __restrict__ pw2,
                        float* __restrict__ t2, int n) {
    int wave = (blockIdx.x * blockDim.x + threadIdx.x) >> 5;
    int lane = threadIdx.x & 31;
    int row0 = wave * 16;
    if (row0 >= n) return;
    int m     = lane & 15;
    int khalf = lane >> 4;

    v16bf a;
    const float* arow = h + (long long)(row0 + m) * HID;
    #pragma unroll
    for (int v = 0; v < 8; ++v) {
        int k = wmma_k(v, khalf);
        a[2 * v]     = (k     < HID) ? (__bf16)arow[k]     : (__bf16)0.f;
        a[2 * v + 1] = (k + 1 < HID) ? (__bf16)arow[k + 1] : (__bf16)0.f;
    }
    const v16bf* bp = (const v16bf*)pw2;

    #pragma unroll
    for (int t = 0; t < 3; ++t) {
        int ncol = t * 16 + m;
        v16bf b = bp[t * 32 + lane];
        v8f acc = {0.f, 0.f, 0.f, 0.f, 0.f, 0.f, 0.f, 0.f};
        acc = __builtin_amdgcn_wmma_f32_16x16x32_bf16(
            false, a, false, b, (short)0, acc, false, false);
        if (ncol < C_OUT) {                    // lane-level store guard only
            #pragma unroll
            for (int v = 0; v < 8; ++v)
                t2[(long long)(row0 + v + 8 * khalf) * C_OUT + ncol] = acc[v];
        }
    }
}

// out = b2 + t2*dinv^2  (self-loop), then scatter2 adds edges on top
__global__ void k_out_init(const float* __restrict__ t2,
                           const float* __restrict__ dinv,
                           const float* __restrict__ b2,
                           float* __restrict__ out, int n) {
    long long i = blockIdx.x * (long long)blockDim.x + threadIdx.x;
    if (i >= (long long)n * C_OUT) return;
    int node = (int)(i / C_OUT);
    int f    = (int)(i % C_OUT);
    float di = dinv[node];
    out[i] = b2[f] + t2[i] * di * di;
}

// ---------------- scatter layer 2: one thread per edge, float4 gather ----------------
__global__ void k_scatter2(const long long* __restrict__ ei,
                           const float* __restrict__ norm,
                           const float* __restrict__ t2,
                           float* __restrict__ out, long long E) {
    long long e = blockIdx.x * (long long)blockDim.x + threadIdx.x;
    if (e >= E) return;
    int src = (int)ei[e];
    int dst = (int)ei[E + e];
    float nw = norm[e];
    const float4* ts = (const float4*)(t2 + (long long)src * C_OUT);  // 160B row, 16B aligned
    float*        od = out + (long long)dst * C_OUT;
    #pragma unroll
    for (int q = 0; q < C_OUT / 4; ++q) {
        float4 tv = ts[q];
        fadd(od + 4 * q + 0, tv.x * nw);
        fadd(od + 4 * q + 1, tv.y * nw);
        fadd(od + 4 * q + 2, tv.z * nw);
        fadd(od + 4 * q + 3, tv.w * nw);
    }
}

extern "C" void kernel_launch(void* const* d_in, const int* in_sizes, int n_in,
                              void* d_out, int out_size, void* d_ws, size_t ws_size,
                              hipStream_t stream) {
    const float*     x  = (const float*)d_in[0];
    const long long* ei = (const long long*)d_in[1];   // int64 edge_index (2,E)
    const float*     w  = (const float*)d_in[2];
    const float*     W1 = (const float*)d_in[3];
    const float*     b1 = (const float*)d_in[4];
    const float*     W2 = (const float*)d_in[5];
    const float*     b2 = (const float*)d_in[6];
    float*           out = (float*)d_out;

    int       n = in_sizes[0] / F_IN;
    long long E = (long long)in_sizes[1] / 2;

    // workspace: pw1(bf16) | pw2(bf16) | deg/dinv | norm | h1 | agg1 | t2
    __bf16* pw1 = (__bf16*)d_ws;                       // 8*32*16 = 4096 bf16 (8KB)
    __bf16* pw2 = pw1 + 8 * 32 * 16;                   // 3*32*16 = 1536 bf16 (3KB)
    float*  deg  = (float*)(pw2 + 3 * 32 * 16);        // 32B-aligned region ends here
    float*  norm = deg  + n;
    float*  h1   = norm + E;
    float*  agg1 = h1   + (long long)n * HID;
    float*  t2   = agg1 + (long long)n * HID;

    const int B = 256;
    unsigned gN  = (unsigned)((n + B - 1) / B);
    unsigned gE  = (unsigned)((E + B - 1) / B);
    long long nh = (long long)n * HID;
    long long nc = (long long)n * C_OUT;
    int tiles    = (n + 15) / 16;
    unsigned gT  = (unsigned)((tiles + 7) / 8);        // 8 waves / 256-thread block

    k_pack_w1  <<<1, 256, 0, stream>>>(W1, pw1);
    k_pack_w2  <<<1, 96, 0, stream>>>(W2, pw2);

    k_init_deg <<<gN, B, 0, stream>>>(deg, n);
    k_deg_accum<<<gE, B, 0, stream>>>(ei, w, deg, E);
    k_dinv     <<<gN, B, 0, stream>>>(deg, n);
    k_norm     <<<gE, B, 0, stream>>>(ei, w, deg, norm, E);

    k_gemm1    <<<gT, 256, 0, stream>>>(x, pw1, h1, n);

    k_zero     <<<(unsigned)((nh + B - 1) / B), B, 0, stream>>>(agg1, nh);
    k_scatter1 <<<gE, B, 0, stream>>>(ei, norm, h1, agg1, E);
    k_relu_self<<<(unsigned)((nh + B - 1) / B), B, 0, stream>>>(h1, deg, b1, agg1, n);

    k_gemm2    <<<gT, 256, 0, stream>>>(agg1, pw2, t2, n);

    k_out_init <<<(unsigned)((nc + B - 1) / B), B, 0, stream>>>(t2, deg, b2, out, n);
    k_scatter2 <<<gE, B, 0, stream>>>(ei, norm, t2, out, E);
}
